// GraphBuildingLSH_13477607375225
// MI455X (gfx1250) — compile-verified, matrix operand order checked by reading, not run on hardware
//
#include <hip/hip_runtime.h>
#include <hip/hip_bf16.h>
#include <math.h>

// ---------------- problem constants (fixed by reference) ----------------
#define BATCH     2
#define NPTS      8192
#define DIM       256
#define NBINS     32        // NPTS / BIN_SIZE
#define BINSZ     256
#define TOPK      16
#define PROJ      16        // n_bins/2 projection columns used
#define CBSTRIDE  32        // codebook row stride (256 x 32)

// LDS padding to kill bank conflicts (64 banks x 4B)
#define ROWP   264          // bf16 elements per padded P row (256 + 8)
#define SROWP  260          // floats per padded sim-stripe row  (256 + 4), %4==0 for float4

typedef __bf16 bf16_t;
typedef __attribute__((ext_vector_type(16))) __bf16 v16bf;
typedef __attribute__((ext_vector_type(8)))  __bf16 v8bf;
typedef __attribute__((ext_vector_type(8)))  float  v8f;

// ======================================================================
// Kernel 1: zero the dense 512MB output (HBM-write bound)
// ======================================================================
__global__ void lsh_zero_out(float4* __restrict__ out, long n4) {
    long i = (long)blockIdx.x * blockDim.x + threadIdx.x;
    long stride = (long)gridDim.x * blockDim.x;
    float4 z = make_float4(0.f, 0.f, 0.f, 0.f);
    for (; i < n4; i += stride) out[i] = z;
}

// ======================================================================
// Kernel 2: bin assignment  (proj = x @ cb[:, :16]; argmax over [p, -p])
// ======================================================================
__global__ __launch_bounds__(256) void lsh_bin_assign(
    const float* __restrict__ x, const float* __restrict__ cb,
    int* __restrict__ bin_idx) {
    int gid = blockIdx.x * blockDim.x + threadIdx.x;   // 0 .. B*N-1
    const float* xr = x + (size_t)gid * DIM;
    float acc[PROJ];
#pragma unroll
    for (int j = 0; j < PROJ; ++j) acc[j] = 0.f;
    for (int d = 0; d < DIM; ++d) {
        float xv = xr[d];
        const float* cr = cb + d * CBSTRIDE;
#pragma unroll
        for (int j = 0; j < PROJ; ++j) acc[j] += xv * cr[j];
    }
    // first-index-wins argmax over [acc, -acc]  (matches jnp.argmax)
    float best = acc[0]; int bi = 0;
#pragma unroll
    for (int j = 1; j < PROJ; ++j) { if (acc[j] > best) { best = acc[j]; bi = j; } }
#pragma unroll
    for (int j = 0; j < PROJ; ++j) { float v = -acc[j]; if (v > best) { best = v; bi = PROJ + j; } }
    bin_idx[gid] = bi;
}

// ======================================================================
// Kernel 3: stable counting sort per batch -> order[] == stable argsort
// one block of 256 threads per batch; 32 elements per thread chunk
// ======================================================================
__global__ __launch_bounds__(256) void lsh_stable_sort(
    const int* __restrict__ bin_idx, int* __restrict__ order) {
    __shared__ int hist[NBINS * 256];   // hist[bin*256 + tid]
    __shared__ int binTotal[NBINS];
    __shared__ int binStart[NBINS];
    int b = blockIdx.x, tid = threadIdx.x;
    const int* keys = bin_idx + b * NPTS;
    const int CHUNK = NPTS / 256;       // 32

    for (int bin = 0; bin < NBINS; ++bin) hist[bin * 256 + tid] = 0;
    __syncthreads();

    int base = tid * CHUNK;
    for (int i = 0; i < CHUNK; ++i) {
        int k = keys[base + i];
        hist[k * 256 + tid] += 1;       // exclusive slot, no atomic needed
    }
    __syncthreads();

    if (tid < NBINS) {                  // per-bin exclusive scan over chunks
        int run = 0;
        for (int t = 0; t < 256; ++t) {
            int v = hist[tid * 256 + t];
            hist[tid * 256 + t] = run;
            run += v;
        }
        binTotal[tid] = run;
    }
    __syncthreads();
    if (tid == 0) {                     // exclusive scan over bins
        int run = 0;
        for (int bin = 0; bin < NBINS; ++bin) { binStart[bin] = run; run += binTotal[bin]; }
    }
    __syncthreads();

    for (int i = 0; i < CHUNK; ++i) {   // stable scatter
        int n = base + i;
        int k = keys[n];
        int pos = binStart[k] + hist[k * 256 + tid];
        hist[k * 256 + tid] += 1;
        order[b * NPTS + pos] = n;
    }
}

// ======================================================================
// Kernel 4: per-bin  sigmoid(P P^T)  via bf16 WMMA + top-16 + scatter
// grid = B*NBINS blocks, 256 threads (8 wave32)
// dynamic LDS: P bf16 [256][ROWP] | 8 x stripe fp32 [16][SROWP] | ordG[256]
// ======================================================================
#define P_BYTES   (256 * ROWP * 2)                 // 135168
#define S_BYTES   (8 * 16 * SROWP * 4)             // 133120
#define O_BYTES   (256 * 4)                        // 1024
#define SMEM_TOTAL (P_BYTES + S_BYTES + O_BYTES)   // 269312 <= 320KB

// load a 16x32 bf16 fragment (A layout; B uses the mirrored N<->M role)
__device__ inline v16bf load_frag(const bf16_t* rowbase, int lane, int kc) {
    int ksel = (lane >> 4) * 8;                    // lanes 0-15: K 0..7/16..23, 16-31: 8..15/24..31
    const bf16_t* p = rowbase + kc * 32 + ksel;
    v8bf lo = *(const v8bf*)(p);
    v8bf hi = *(const v8bf*)(p + 16);
    return __builtin_shufflevector(lo, hi, 0, 1, 2, 3, 4, 5, 6, 7,
                                           8, 9, 10, 11, 12, 13, 14, 15);
}

__global__ __launch_bounds__(256) void lsh_bin_sim_topk(
    const float* __restrict__ x, const int* __restrict__ order,
    float* __restrict__ out) {
    extern __shared__ char smem[];
    bf16_t* Pl   = (bf16_t*)smem;
    float*  Sl   = (float*)(smem + P_BYTES);
    int*    ordG = (int*)(smem + P_BYTES + S_BYTES);

    int bb  = blockIdx.x;
    int b   = bb >> 5;          // batch
    int bin = bb & 31;
    int tid = threadIdx.x;
    int lane = tid & 31;
    int w    = tid >> 5;        // wave id 0..7

    // cache this bin's slice of the permutation
    ordG[tid] = order[b * NPTS + bin * BINSZ + tid];
    __syncthreads();

    // stage P (gathered rows) into LDS as bf16; thread t loads row t
    {
        const float4* src = (const float4*)(x + ((size_t)b * NPTS + ordG[tid]) * DIM);
        bf16_t* dst = Pl + tid * ROWP;
#pragma unroll 4
        for (int i = 0; i < DIM / 4; ++i) {
            float4 v = src[i];
            dst[4 * i + 0] = (bf16_t)v.x;
            dst[4 * i + 1] = (bf16_t)v.y;
            dst[4 * i + 2] = (bf16_t)v.z;
            dst[4 * i + 3] = (bf16_t)v.w;
        }
    }
    __syncthreads();

    float* Sw = Sl + w * 16 * SROWP;   // this wave's private 16x256 sim stripe
    int rsel = lane & 15;
    int m0   = (lane < 16) ? 0 : 8;
    int ncol = lane & 15;

    for (int sp = w; sp < 16; sp += 8) {           // two stripes per wave
        int rowBase = sp * 16;
        const bf16_t* arow = Pl + (rowBase + rsel) * ROWP;

        v16bf afrag[8];
#pragma unroll
        for (int kc = 0; kc < 8; ++kc) afrag[kc] = load_frag(arow, lane, kc);

        for (int j = 0; j < 16; ++j) {             // 16 column blocks
            const bf16_t* brow = Pl + (j * 16 + rsel) * ROWP;

            // preload the whole 32x16 (K=256) B tile -> one dscnt wait,
            // then 8 back-to-back WMMAs chained only through the accumulator
            v16bf bfrag[8];
#pragma unroll
            for (int kc = 0; kc < 8; ++kc) bfrag[kc] = load_frag(brow, lane, kc);

            v8f acc = {};
#pragma unroll
            for (int kc = 0; kc < 8; ++kc) {       // K = 256 in chunks of 32
                acc = __builtin_amdgcn_wmma_f32_16x16x32_bf16(
                    false, afrag[kc], false, bfrag[kc], (short)0, acc, false, false);
            }

            // sigmoid via fast rcp (v_exp + v_add + v_rcp, no IEEE div chain)
#pragma unroll
            for (int r8 = 0; r8 < 8; ++r8) {
                float e = __expf(-acc[r8]);
                float s = __builtin_amdgcn_rcpf(1.0f + e);
                Sw[(r8 + m0) * SROWP + (j * 16 + ncol)] = s;
            }
        }

        // top-16 per row: iterative max-extract with float4 LDS scans
        // (first-index tie-break via strict > in element order, matches lax.top_k);
        // values in (0,1) so negative sentinels are safe.
        if (lane < 16) {
            int row = lane;
            float* srow = Sw + row * SROWP;
            const float4* srow4 = (const float4*)srow;
            int gsrc = ordG[rowBase + row];
            float* orow = out + ((size_t)b * NPTS + gsrc) * NPTS;
#pragma unroll 1
            for (int p = 0; p < TOPK; ++p) {
                float best = -2.0f; int bi = 0;
#pragma unroll 1
                for (int n4 = 0; n4 < BINSZ / 4; ++n4) {
                    float4 v = srow4[n4];
                    int nb = n4 * 4;
                    if (v.x > best) { best = v.x; bi = nb; }
                    if (v.y > best) { best = v.y; bi = nb + 1; }
                    if (v.z > best) { best = v.z; bi = nb + 2; }
                    if (v.w > best) { best = v.w; bi = nb + 3; }
                }
                srow[bi] = -3.0f;
                orow[ordG[bi]] = best;   // permutation => no collisions, plain store
            }
        }
        // Sw is wave-private; no __syncthreads needed between stripes
    }
}

// ======================================================================
// launcher
// ======================================================================
extern "C" void kernel_launch(void* const* d_in, const int* in_sizes, int n_in,
                              void* d_out, int out_size, void* d_ws, size_t ws_size,
                              hipStream_t stream) {
    const float* x  = (const float*)d_in[0];   // (2, 8192, 256) fp32
    const float* cb = (const float*)d_in[1];   // (256, 32) fp32
    float* out = (float*)d_out;                // (2, 8192, 8192) fp32

    int* bin_idx = (int*)d_ws;                               // B*N ints
    int* order   = (int*)((char*)d_ws + BATCH * NPTS * 4);   // B*N ints

    // 1) zero the 512MB output
    long n4 = (long)BATCH * NPTS * NPTS / 4;
    lsh_zero_out<<<8192, 256, 0, stream>>>((float4*)out, n4);

    // 2) bin ids
    lsh_bin_assign<<<(BATCH * NPTS) / 256, 256, 0, stream>>>(x, cb, bin_idx);

    // 3) stable argsort by bin id (counting sort)
    lsh_stable_sort<<<BATCH, 256, 0, stream>>>(bin_idx, order);

    // 4) per-bin WMMA similarity + sigmoid + top-16 + scatter
    lsh_bin_sim_topk<<<BATCH * NBINS, 256, SMEM_TOTAL, stream>>>(x, order, out);
}
